// DARQNModel_48962627174732
// MI455X (gfx1250) — matrix operand order repaired and checked by reading, not compiled, not run on hardware
//
#include <hip/hip_runtime.h>

#define T_FR 2048
#define HID  256
#define NACT 18
#define NL   2

typedef unsigned short ushort_t;
typedef unsigned int   uint_t;
typedef __attribute__((ext_vector_type(16))) __bf16 v16bf;
typedef __attribute__((ext_vector_type(8)))  float  v8f;
typedef __attribute__((ext_vector_type(4)))  unsigned int u32x4;
typedef __attribute__((ext_vector_type(8)))  unsigned int u32x8;

// ---------------- bf16 helpers ----------------
__device__ __forceinline__ ushort_t f2bf(float f) {
  union { float f; uint_t u; } v; v.f = f;
  uint_t r = v.u + 0x7FFFu + ((v.u >> 16) & 1u);
  return (ushort_t)(r >> 16);
}
__device__ __forceinline__ float bf2f(ushort_t h) {
  union { uint_t u; float f; } v; v.u = ((uint_t)h) << 16; return v.f;
}
__device__ __forceinline__ __bf16 bfbits(ushort_t h) {
  union { ushort_t u; __bf16 b; } v; v.u = h; return v.b;
}
__device__ __forceinline__ float bflo(uint_t p) {
  union { uint_t u; float f; } v; v.u = p << 16; return v.f;
}
__device__ __forceinline__ float bfhi(uint_t p) {
  union { uint_t u; float f; } v; v.u = p & 0xFFFF0000u; return v.f;
}
__device__ __forceinline__ float sigm_(float x) { return 1.0f / (1.0f + __expf(-x)); }

// ---------------- CDNA5 async data movers ----------------
// Per-lane async global->LDS copy, 16B per lane (ISA 15.18.3 opc 98), ASYNCcnt.
__device__ __forceinline__ void async_b128(unsigned lds_off, const void* g) {
  asm volatile("global_load_async_to_lds_b128 %0, %1, off"
               :: "v"(lds_off), "v"(g) : "memory");
}
__device__ __forceinline__ void wait_async0() {
  asm volatile("s_wait_asynccnt 0" ::: "memory");
}

// Tensor Data Mover: 2D tile 49x256 (bf16) global->LDS with hardware row
// padding of 4 DWORDs after every 128 DWORDs (reproduces R_LDA=264 layout).
// D# built per ISA 8.3/8.4; groups 2/3 zero (2D tile, tile_dim2=0 unused).
__device__ __forceinline__ void tdm_load_tile49x256(unsigned lds_off, const ushort_t* gsrc) {
  unsigned long long ga = (unsigned long long)(uintptr_t)gsrc;
  u32x4 g0;
  g0.x = 1u;                                                  // count=1, user mode
  g0.y = lds_off;                                             // lds_addr [63:32]
  g0.z = (unsigned)ga;                                        // global_addr [95:64]
  g0.w = ((unsigned)(ga >> 32) & 0x01FFFFFFu) | (2u << 30);   // [120:96] | type=2
  u32x8 g1;
  g1[0] = (1u << 16)       // data_size = 2B
        | (1u << 20)       // pad_enable
        | (6u << 22)       // pad_interval: 2^(6+1)=128 DWORDs (one 512B row)
        | (3u << 25);      // pad_amount: 4 DWORDs (16B)
  g1[1] = (256u << 16);    // tensor_dim0 = 256
  g1[2] = (49u  << 16);    // tensor_dim1 = 49
  g1[3] = (256u << 16);    // tile_dim0 = 256
  g1[4] = 49u;             // tile_dim1 = 49 (tile_dim2 = 0)
  g1[5] = 256u;            // tensor_dim0_stride = 256
  g1[6] = 0u;
  g1[7] = 0u;
  u32x4 gz; gz.x = 0u; gz.y = 0u; gz.z = 0u; gz.w = 0u;
  asm volatile("tensor_load_to_lds %0, %1, %2, %3"
               :: "s"(g0), "s"(g1), "s"(gz), "s"(gz) : "memory");
}

// ---------------- WMMA fragment loaders (ISA 7.12.2 layouts, wave32) ------
__device__ __forceinline__ v16bf load_afrag(const ushort_t* base, int stride,
                                            int m0, int k0, int lane, int mclamp) {
  int m = m0 + (lane & 15);
  if (m > mclamp) m = mclamp;
  int kb = k0 + ((lane & 16) ? 8 : 0);
  const ushort_t* p = base + m * stride + kb;
  v16bf a;
#pragma unroll
  for (int v = 0; v < 8; ++v) {
    int k = ((v & 4) ? 16 : 0) + (v & 3) * 2;
    a[2 * v]     = bfbits(p[k]);
    a[2 * v + 1] = bfbits(p[k + 1]);
  }
  return a;
}
__device__ __forceinline__ v16bf load_bfrag(const ushort_t* B, int ldn,
                                            int k0, int n0, int lane) {
  int kb = k0 + ((lane & 16) ? 8 : 0);
  const ushort_t* p = B + (size_t)kb * ldn + n0 + (lane & 15);
  v16bf b;
#pragma unroll
  for (int e = 0; e < 16; ++e) b[e] = bfbits(p[(size_t)(((e & 8) ? 16 : 0) + (e & 7)) * ldn]);
  return b;
}
#define WMMA_BF16(a, b, c) \
  __builtin_amdgcn_wmma_f32_16x16x32_bf16(false, (a), false, (b), (short)0, (c), false, false)

// ---------------- weight repack kernels (fp32 -> bf16, GEMM-friendly) -----
__global__ void pack_b3(const float* __restrict__ w, ushort_t* __restrict__ B) {
  int i = blockIdx.x * 256 + threadIdx.x;                // B3[k][n] = w_c3[n][k]
  if (i >= 576 * 256) return;
  int k = i / 256, n = i % 256;
  B[i] = f2bf(w[n * 576 + k]);
}
__global__ void pack_t256(const float* __restrict__ w, ushort_t* __restrict__ B) {
  int i = blockIdx.x * 256 + threadIdx.x;                // Ba[k][n] = w_a[n][k]
  if (i >= 256 * 256) return;
  int k = i / 256, n = i % 256;
  B[i] = f2bf(w[n * 256 + k]);
}
__global__ void pack_lstm(const float* __restrict__ w, ushort_t* __restrict__ B) {
  int i = blockIdx.x * 256 + threadIdx.x;                // Bw[l][k][n] = w[l][n][k]
  if (i >= NL * 256 * 1024) return;
  int l = i / (256 * 1024), rem = i % (256 * 1024);
  int k = rem / 1024, n = rem % 1024;
  B[i] = f2bf(w[((size_t)l * 1024 + n) * 256 + k]);
}

// ---------------- conv1: 1x84x84 -> 32x20x20 (8x8 stride4) ----------------
__global__ __launch_bounds__(256) void conv1_k(const float* __restrict__ frames,
                                               const float* __restrict__ w,
                                               const float* __restrict__ b,
                                               ushort_t* __restrict__ a1) {
  int t = blockIdx.y;
  int idx = blockIdx.x * 256 + threadIdx.x;
  if (idx >= 32 * 20 * 20) return;
  int c = idx / 400, r = idx % 400, oy = r / 20, ox = r % 20;
  const float* fin = frames + (size_t)t * 84 * 84 + (oy * 4) * 84 + ox * 4;
  const float* wp  = w + c * 64;
  float acc = b[c];
#pragma unroll
  for (int ky = 0; ky < 8; ++ky)
#pragma unroll
    for (int kx = 0; kx < 8; ++kx)
      acc += fin[ky * 84 + kx] * wp[ky * 8 + kx];
  a1[(size_t)t * 12800 + idx] = f2bf(fmaxf(acc, 0.0f));
}

// ---------------- conv2: 32x20x20 -> 64x9x9 (4x4 stride2) -----------------
__global__ __launch_bounds__(256) void conv2_k(const ushort_t* __restrict__ a1,
                                               const float* __restrict__ w,
                                               const float* __restrict__ b,
                                               ushort_t* __restrict__ a2) {
  int t = blockIdx.y;
  int idx = blockIdx.x * 256 + threadIdx.x;
  if (idx >= 64 * 81) return;
  int c2 = idx / 81, r = idx % 81, oy = r / 9, ox = r % 9;
  const ushort_t* in = a1 + (size_t)t * 12800;
  float acc = b[c2];
  for (int ci = 0; ci < 32; ++ci) {
    const ushort_t* ip = in + ci * 400 + (oy * 2) * 20 + ox * 2;
    const float* wp = w + ((c2 * 32 + ci) * 16);
#pragma unroll
    for (int ky = 0; ky < 4; ++ky)
#pragma unroll
      for (int kx = 0; kx < 4; ++kx)
        acc += bf2f(ip[ky * 20 + kx]) * wp[ky * 4 + kx];
  }
  a2[(size_t)t * 5184 + idx] = f2bf(fmaxf(acc, 0.0f));
}

// ---------------- conv3 as WMMA implicit GEMM: per frame [49,576]x[576,256]
// B3 k-slabs (32x256) are async-staged into double-buffered LDS (one copy
// shared by all 8 waves instead of 8x redundant global reads), overlapped
// with WMMA compute of the previous slab.
#define C3_LDA 296   // 288 k-slab + 8 pad
#define C3_LDB 264   // 256 + 8 pad (B slab rows)
__global__ __launch_bounds__(256) void conv3_wmma(const ushort_t* __restrict__ a2,
                                                  const ushort_t* __restrict__ B3,
                                                  const float* __restrict__ b3,
                                                  ushort_t* __restrict__ vecs) {
  __shared__ ushort_t As[64 * C3_LDA];         // 37,888 B
  __shared__ ushort_t Bs[2][32 * C3_LDB];      // 2 x 16,896 B
  int t = blockIdx.x;
  const ushort_t* in = a2 + (size_t)t * 5184;
  int tid = threadIdx.x, wave = tid >> 5, lane = tid & 31, hi = (lane >> 4) & 1;

  // async-stage k-slab g (rows [g*32, g*32+32) of B3) into buffer b
  auto stage_B = [&](int g, int b) {
    const ushort_t* src = B3 + (size_t)g * 32 * 256;
    unsigned base = (unsigned)(uintptr_t)&Bs[b][0];
    for (int c = tid; c < 1024; c += 256) {       // 16B per chunk, 4 per thread
      int rr = c >> 5, cc = c & 31;
      async_b128(base + (unsigned)(rr * C3_LDB + cc * 8) * 2, src + rr * 256 + cc * 8);
    }
  };

  v8f acc[2][4] = {};                          // ntiles {2w,2w+1} x mtiles 0..3
  stage_B(0, 0);
  for (int slab = 0; slab < 2; ++slab) {
    if (slab) __syncthreads();                 // protect As rewrite
    for (int i = tid; i < 64 * 288; i += 256) {      // im2col fill (pad rows = 0)
      int m = i / 288, kk = i % 288;
      ushort_t v = 0;
      if (m < 49) {
        int k = slab * 288 + kk;
        int oy = m / 7, ox = m % 7, c = k / 9, rr = (k % 9) / 3, ss = k % 3;
        v = in[c * 81 + (oy + rr) * 9 + (ox + ss)];
      }
      As[m * C3_LDA + kk] = v;
    }
    __syncthreads();
    for (int ks = 0; ks < 9; ++ks) {
      int g = slab * 9 + ks;
      wait_async0();                           // slab g landed (this wave's chunks)
      __syncthreads();                         // ... and everyone else's
      if (g + 1 < 18) stage_B(g + 1, (g + 1) & 1);   // overlap next slab DMA
      int k0 = ks * 32;
      const ushort_t* Bsl = &Bs[g & 1][0];
      v16bf af[4];
#pragma unroll
      for (int mt = 0; mt < 4; ++mt) af[mt] = load_afrag(As, C3_LDA, mt * 16, k0, lane, 63);
#pragma unroll
      for (int j = 0; j < 2; ++j) {
        v16bf bf = load_bfrag(Bsl, C3_LDB, 0, (wave * 2 + j) * 16, lane);
#pragma unroll
        for (int mt = 0; mt < 4; ++mt) acc[j][mt] = WMMA_BF16(af[mt], bf, acc[j][mt]);
      }
    }
  }
#pragma unroll
  for (int j = 0; j < 2; ++j) {
    int col = (wave * 2 + j) * 16 + (lane & 15);
    float bb = b3[col];
#pragma unroll
    for (int mt = 0; mt < 4; ++mt)
#pragma unroll
      for (int r = 0; r < 8; ++r) {
        int row = mt * 16 + hi * 8 + r;
        if (row < 49)
          vecs[(size_t)t * 49 * 256 + row * 256 + col] = f2bf(fmaxf(acc[j][mt][r] + bb, 0.0f));
      }
  }
}

// ---------------- persistent recurrent kernel: 1 WG, 512 threads, 16 waves
// V tiles are TDM double-buffered: wave 0 issues tensor_load_to_lds for tile
// t+1 at the top of step t, hiding the entire tile latency behind the step.
#define R_LDA 264    // 256 + 8 pad (matches TDM pad_interval/amount)
__global__ __launch_bounds__(512) void recurrent_k(const ushort_t* __restrict__ vecs,
                                                   const ushort_t* __restrict__ Ba1,
                                                   const ushort_t* __restrict__ Ba2,
                                                   const float* __restrict__ b_a1,
                                                   const float* __restrict__ b_a2,
                                                   const uint_t* __restrict__ Wih,
                                                   const uint_t* __restrict__ Whh,
                                                   const float* __restrict__ b_ih,
                                                   const float* __restrict__ b_hh,
                                                   const float* __restrict__ w_q,
                                                   const float* __restrict__ b_q,
                                                   float* __restrict__ out) {
  __shared__ ushort_t sV[2][49 * R_LDA];   // 2 x 25,872 B (TDM double buffer)
  __shared__ ushort_t sE[49 * R_LDA];      // 25,872 B
  __shared__ float pmax[49 * 4], psum[49 * 4];
  __shared__ float rmax[49], rinv[49];
  __shared__ float sX[256];
  __shared__ float sH[NL * 256], sC[NL * 256], sG[1024];

  int tid = threadIdx.x, wave = tid >> 5, lane = tid & 31, hi = (lane >> 4) & 1;
  int mt = wave & 3, ntg = wave >> 2;
  if (tid < NL * 256) { sH[tid] = 0.0f; sC[tid] = 0.0f; }

  unsigned sv_off0 = (unsigned)(uintptr_t)&sV[0][0];
  unsigned sv_off1 = (unsigned)(uintptr_t)&sV[1][0];
  if (tid < 32) tdm_load_tile49x256(sv_off0, vecs);          // tile 0 in flight
  __syncthreads();

  for (int t = 0; t < T_FR; ++t) {
    const ushort_t* sVb = &sV[t & 1][0];
    if (tid < 32) {
      __builtin_amdgcn_s_wait_tensorcnt(0);                  // tile t landed
      if (t + 1 < T_FR)                                      // launch tile t+1 DMA
        tdm_load_tile49x256((t & 1) ? sv_off0 : sv_off1,
                            vecs + (size_t)(t + 1) * 49 * 256);
    }
    __syncthreads();                                         // B1: tile t visible

    // GEMM1: E = tanh(V @ Ba1 + b_a1 + h1)
    {
      v8f acc[4] = {};
      for (int ks = 0; ks < 8; ++ks) {
        int k0 = ks * 32;
        v16bf af = load_afrag(sVb, R_LDA, mt * 16, k0, lane, 48);
#pragma unroll
        for (int j = 0; j < 4; ++j) {
          v16bf bf = load_bfrag(Ba1, 256, k0, ntg * 64 + j * 16, lane);
          acc[j] = WMMA_BF16(af, bf, acc[j]);
        }
      }
#pragma unroll
      for (int j = 0; j < 4; ++j) {
        int col = ntg * 64 + j * 16 + (lane & 15);
        float hb = b_a1[col] + sH[256 + col];
#pragma unroll
        for (int r = 0; r < 8; ++r) {
          int row = mt * 16 + hi * 8 + r;
          if (row < 49) sE[row * R_LDA + col] = f2bf(tanhf(acc[j][r] + hb));
        }
      }
    }
    __syncthreads();                                         // B2

    // GEMM2: F = E @ Ba2 + b_a2 (kept in registers), two-pass softmax + ctx
    v8f acc[4] = {};
    for (int ks = 0; ks < 8; ++ks) {
      int k0 = ks * 32;
      v16bf af = load_afrag(sE, R_LDA, mt * 16, k0, lane, 48);
#pragma unroll
      for (int j = 0; j < 4; ++j) {
        v16bf bf = load_bfrag(Ba2, 256, k0, ntg * 64 + j * 16, lane);
        acc[j] = WMMA_BF16(af, bf, acc[j]);
      }
    }
    float b2v[4];
#pragma unroll
    for (int j = 0; j < 4; ++j) b2v[j] = b_a2[ntg * 64 + j * 16 + (lane & 15)];
#pragma unroll
    for (int r = 0; r < 8; ++r) {
      int row = mt * 16 + hi * 8 + r;
      if (row < 49) {
        float mx = -3.0e38f;
        float ev[4];
#pragma unroll
        for (int j = 0; j < 4; ++j) { ev[j] = acc[j][r] + b2v[j]; mx = fmaxf(mx, ev[j]); }
        for (int off = 1; off < 16; off <<= 1) mx = fmaxf(mx, __shfl_xor(mx, off));
        float sm = 0.0f;
#pragma unroll
        for (int j = 0; j < 4; ++j) sm += __expf(ev[j] - mx);
        for (int off = 1; off < 16; off <<= 1) sm += __shfl_xor(sm, off);
        if ((lane & 15) == 0) { pmax[row * 4 + ntg] = mx; psum[row * 4 + ntg] = sm; }
      }
    }
    if (tid < 256) sX[tid] = 0.0f;
    __syncthreads();                                         // B3
    if (tid < 49) {
      float gm = -3.0e38f;
#pragma unroll
      for (int i = 0; i < 4; ++i) gm = fmaxf(gm, pmax[tid * 4 + i]);
      float gs = 0.0f;
#pragma unroll
      for (int i = 0; i < 4; ++i) gs += psum[tid * 4 + i] * __expf(pmax[tid * 4 + i] - gm);
      rmax[tid] = gm; rinv[tid] = 1.0f / gs;
    }
    __syncthreads();                                         // B4
    {
      float cacc[4] = {0.0f, 0.0f, 0.0f, 0.0f};
#pragma unroll
      for (int r = 0; r < 8; ++r) {
        int row = mt * 16 + hi * 8 + r;
        if (row < 49) {
          float rm = rmax[row], ri = rinv[row];
#pragma unroll
          for (int j = 0; j < 4; ++j) {
            int col = ntg * 64 + j * 16 + (lane & 15);
            float s = __expf(acc[j][r] + b2v[j] - rm) * ri;
            cacc[j] += s * bf2f(sVb[row * R_LDA + col]);
          }
        }
      }
#pragma unroll
      for (int j = 0; j < 4; ++j)
        atomicAdd(&sX[ntg * 64 + j * 16 + (lane & 15)], cacc[j]);
    }
    __syncthreads();                                         // B5

    // stacked LSTM, gemv over L2-resident bf16 weights [l][k][n]
    const float* xin = sX;
    for (int l = 0; l < NL; ++l) {
      int n0 = tid * 2;
      float a0  = b_ih[l * 1024 + n0]     + b_hh[l * 1024 + n0];
      float a1v = b_ih[l * 1024 + n0 + 1] + b_hh[l * 1024 + n0 + 1];
      const uint_t* Wi = Wih + (size_t)l * 131072 + tid;
      const uint_t* Wh = Whh + (size_t)l * 131072 + tid;
      const float* hin = sH + l * 256;
#pragma unroll 4
      for (int k = 0; k < 256; ++k) {
        uint_t wp = Wi[k * 512];
        uint_t hp = Wh[k * 512];
        float xv = xin[k], hv = hin[k];
        a0  += xv * bflo(wp) + hv * bflo(hp);
        a1v += xv * bfhi(wp) + hv * bfhi(hp);
      }
      sG[n0] = a0; sG[n0 + 1] = a1v;
      __syncthreads();
      if (tid < 256) {
        int j = tid;
        float ig = sG[j], fg = sG[256 + j], gg = sG[512 + j], og = sG[768 + j];
        float cn = sigm_(fg) * sC[l * 256 + j] + sigm_(ig) * tanhf(gg);
        float hn = sigm_(og) * tanhf(cn);
        sC[l * 256 + j] = cn; sH[l * 256 + j] = hn;
      }
      __syncthreads();
      xin = sH;
    }

    if (t == T_FR - 1 && tid < NACT) {
      float a = b_q[tid];
      const float* wq = w_q + tid * 256;
      for (int k = 0; k < 256; ++k) a += sH[256 + k] * wq[k];
      out[tid] = a;
    }
  }
}

// ---------------- host side ----------------
extern "C" void kernel_launch(void* const* d_in, const int* in_sizes, int n_in,
                              void* d_out, int out_size, void* d_ws, size_t ws_size,
                              hipStream_t stream) {
  (void)in_sizes; (void)n_in; (void)out_size; (void)ws_size;
  const float* frames = (const float*)d_in[0];
  const float* w_c1 = (const float*)d_in[1];  const float* b_c1 = (const float*)d_in[2];
  const float* w_c2 = (const float*)d_in[3];  const float* b_c2 = (const float*)d_in[4];
  const float* w_c3 = (const float*)d_in[5];  const float* b_c3 = (const float*)d_in[6];
  const float* w_a1 = (const float*)d_in[7];  const float* b_a1 = (const float*)d_in[8];
  const float* w_a2 = (const float*)d_in[9];  const float* b_a2 = (const float*)d_in[10];
  const float* w_ih = (const float*)d_in[11]; const float* w_hh = (const float*)d_in[12];
  const float* b_ih = (const float*)d_in[13]; const float* b_hh = (const float*)d_in[14];
  const float* w_q  = (const float*)d_in[15]; const float* b_q  = (const float*)d_in[16];

  char* ws = (char*)d_ws;
  size_t off = 0;
  auto take = [&](size_t bytes) -> void* {
    void* p = ws + off;
    off += (bytes + 255) & ~(size_t)255;
    return p;
  };
  ushort_t* a1   = (ushort_t*)take((size_t)T_FR * 32 * 400 * 2);
  ushort_t* a2   = (ushort_t*)take((size_t)T_FR * 64 * 81  * 2);
  ushort_t* vecs = (ushort_t*)take((size_t)T_FR * 49 * 256 * 2);
  ushort_t* B3   = (ushort_t*)take((size_t)576 * 256 * 2);
  ushort_t* Ba1  = (ushort_t*)take((size_t)256 * 256 * 2);
  ushort_t* Ba2  = (ushort_t*)take((size_t)256 * 256 * 2);
  ushort_t* Wih  = (ushort_t*)take((size_t)NL * 256 * 1024 * 2);
  ushort_t* Whh  = (ushort_t*)take((size_t)NL * 256 * 1024 * 2);

  pack_b3  <<<(576 * 256 + 255) / 256, 256, 0, stream>>>(w_c3, B3);
  pack_t256<<<256, 256, 0, stream>>>(w_a1, Ba1);
  pack_t256<<<256, 256, 0, stream>>>(w_a2, Ba2);
  pack_lstm<<<(NL * 256 * 1024 + 255) / 256, 256, 0, stream>>>(w_ih, Wih);
  pack_lstm<<<(NL * 256 * 1024 + 255) / 256, 256, 0, stream>>>(w_hh, Whh);

  conv1_k<<<dim3(50, T_FR), 256, 0, stream>>>(frames, w_c1, b_c1, a1);
  conv2_k<<<dim3(21, T_FR), 256, 0, stream>>>(a1, w_c2, b_c2, a2);
  conv3_wmma<<<T_FR, 256, 0, stream>>>(a2, B3, b_c3, vecs);

  recurrent_k<<<1, 512, 0, stream>>>(vecs, Ba1, Ba2, b_a1, b_a2,
                                     (const uint_t*)Wih, (const uint_t*)Whh,
                                     b_ih, b_hh, w_q, b_q, (float*)d_out);
}
